// AttnBlock_91345364451336
// MI455X (gfx1250) — matrix operand order repaired and checked by reading, not compiled, not run on hardware
//
#include <hip/hip_runtime.h>
#include <hip/hip_bf16.h>

typedef __attribute__((ext_vector_type(16))) _Float16 v16h;
typedef __attribute__((ext_vector_type(8)))  _Float16 v8h;
typedef __attribute__((ext_vector_type(8)))  float    v8f;
typedef __attribute__((ext_vector_type(4)))  float    v4f;

#define NPIX 4096
#define CCH  256

// ---------- WMMA fragment helpers (contiguous k order, same for A and B) ----------
// A (16x32 f16): lane m=lane&15; halves t: k = t + 16*(lane>=16)
// -> two b128 loads from a row-major (M, K) matrix.
__device__ __forceinline__ v16h load_a_frag(const _Float16* __restrict__ base,
                                            int ld, int k0, int lane) {
    int m  = lane & 15;
    int hb = (lane >> 4) << 4;            // 0 or 16
    const _Float16* p = base + (size_t)m * ld + (k0 + hb);
    v8h lo = *(const v8h*)(p);
    v8h hi = *(const v8h*)(p + 8);
    v16h r;
#pragma unroll
    for (int i = 0; i < 8; ++i) { r[i] = lo[i]; r[i + 8] = hi[i]; }
    return r;
}

// B (32x16 f16): lane n=lane&15; halves t: k = t + 16*(lane>=16)
// -> two b128 loads from a (N, K) matrix (k contiguous per row).
__device__ __forceinline__ v16h load_b_frag(const _Float16* __restrict__ base,
                                            int ld, int k0, int lane) {
    int n  = lane & 15;
    int hb = (lane >> 4) << 4;            // 0 or 16
    const _Float16* p = base + (size_t)n * ld + (k0 + hb);
    v8h lo = *(const v8h*)(p);
    v8h hi = *(const v8h*)(p + 8);
    v16h r;
#pragma unroll
    for (int i = 0; i < 8; ++i) { r[i] = lo[i]; r[i + 8] = hi[i]; }
    return r;
}

__device__ __forceinline__ v8f wmma_f16(v16h a, v16h b, v8f c) {
    return __builtin_amdgcn_wmma_f32_16x16x32_f16(false, a, false, b, (short)0, c,
                                                  false, false);
}

// ---------- GroupNorm: (b,c,h,w) f32 -> h16 (b, n, c) f16 ----------
__global__ void gn_kernel(const float* __restrict__ x, const float* __restrict__ gamma,
                          const float* __restrict__ beta, _Float16* __restrict__ h16) {
    int g = blockIdx.x, b = blockIdx.y;
    int t = threadIdx.x;
    int c0 = g * 8;
    const float* reg = x + ((size_t)b * CCH + c0) * NPIX;
    float s = 0.f, ss = 0.f;
    for (int i = 0; i < 32; ++i) {
        v4f v = *(const v4f*)(reg + t * 4 + i * 1024);
#pragma unroll
        for (int j = 0; j < 4; ++j) { s += v[j]; ss += v[j] * v[j]; }
    }
#pragma unroll
    for (int off = 16; off >= 1; off >>= 1) {
        s  += __shfl_xor(s, off, 32);
        ss += __shfl_xor(ss, off, 32);
    }
    __shared__ float sh_s[8], sh_ss[8];
    int wave = t >> 5, lane = t & 31;
    if (lane == 0) { sh_s[wave] = s; sh_ss[wave] = ss; }
    __syncthreads();
    float ts = 0.f, tss = 0.f;
#pragma unroll
    for (int i = 0; i < 8; ++i) { ts += sh_s[i]; tss += sh_ss[i]; }
    const float invn = 1.0f / 32768.0f;
    float mean = ts * invn;
    float var  = tss * invn - mean * mean;
    float rstd = rsqrtf(var + 1e-6f);
#pragma unroll
    for (int ci = 0; ci < 8; ++ci) {
        int c = c0 + ci;
        float gm = gamma[c] * rstd;
        float bt = beta[c] - mean * gm;
        const float* row = reg + (size_t)ci * NPIX;
        _Float16* hh = h16 + (size_t)b * NPIX * CCH + c;
        for (int i = 0; i < 4; ++i) {
            int n0 = t * 4 + i * 1024;
            v4f v = *(const v4f*)(row + n0);
#pragma unroll
            for (int j = 0; j < 4; ++j)
                hh[(size_t)(n0 + j) * CCH] = (_Float16)(v[j] * gm + bt);
        }
    }
}

// ---------- weights f32 -> f16 ----------
// wq pre-scaled by C^-0.5 * log2(e) so attention scores live in the exp2 domain:
// softmax(s*scale) == exp2(s*scale*log2e - m2) / sum, with m2 tracked in log2 units.
__global__ void wconv_kernel(const float* __restrict__ wq, const float* __restrict__ wk,
                             const float* __restrict__ wv, const float* __restrict__ wp,
                             _Float16* __restrict__ w16) {
    int i = blockIdx.x * 256 + threadIdx.x;
    int m = i >> 16, off = i & 0xFFFF;
    const float* src = (m == 0) ? wq : (m == 1) ? wk : (m == 2) ? wv : wp;
    float sc = (m == 0) ? (0.0625f * 1.44269504088896340736f) : 1.0f;
    w16[i] = (_Float16)(src[off] * sc);
}

// ---------- GEMM: out[co][n] = sum_ci W[co][ci] * src[n][ci] + bias[co] ----------
// mode 0: f16 out, (n, c) layout      (q, k)
// mode 1: f16 out, (c, n) layout      (v transposed for PV A-fragments)
// mode 2: f32 out, (c, n) + residual  (proj)
__global__ void gemm_kernel(const _Float16* __restrict__ w, const float* __restrict__ bias,
                            const _Float16* __restrict__ src, _Float16* __restrict__ out16,
                            float* __restrict__ outf, const float* __restrict__ xres,
                            int mode) {
    int lane = threadIdx.x & 31, wave = threadIdx.x >> 5;
    int b = blockIdx.z;
    int cobase = (blockIdx.y * 2 + (wave & 1)) * 16;
    int nbase  = blockIdx.x * 256 + (wave >> 1) * 64;
    const _Float16* srcb = src + (size_t)b * NPIX * CCH;

    v8f acc[4];
    v8f zero = {0.f, 0.f, 0.f, 0.f, 0.f, 0.f, 0.f, 0.f};
#pragma unroll
    for (int j = 0; j < 4; ++j) acc[j] = zero;

#pragma unroll
    for (int ks = 0; ks < 8; ++ks) {
        int k0 = ks * 32;
        v16h a = load_a_frag(w + (size_t)cobase * CCH, CCH, k0, lane);
#pragma unroll
        for (int j = 0; j < 4; ++j) {
            v16h bf = load_b_frag(srcb + (size_t)(nbase + j * 16) * CCH, CCH, k0, lane);
            acc[j] = wmma_f16(a, bf, acc[j]);
        }
    }

    int hi = lane >> 4;
    int n0 = nbase + (lane & 15);
    float bv[8];
#pragma unroll
    for (int r = 0; r < 8; ++r) bv[r] = bias[cobase + hi * 8 + r];

#pragma unroll
    for (int j = 0; j < 4; ++j) {
        int n = n0 + j * 16;
        if (mode == 0) {
            v8h pk;
#pragma unroll
            for (int r = 0; r < 8; ++r) pk[r] = (_Float16)(acc[j][r] + bv[r]);
            *(v8h*)(out16 + ((size_t)b * NPIX + n) * CCH + cobase + hi * 8) = pk;
        } else if (mode == 1) {
#pragma unroll
            for (int r = 0; r < 8; ++r) {
                int mrow = cobase + hi * 8 + r;
                out16[((size_t)b * CCH + mrow) * NPIX + n] = (_Float16)(acc[j][r] + bv[r]);
            }
        } else {
#pragma unroll
            for (int r = 0; r < 8; ++r) {
                int mrow = cobase + hi * 8 + r;
                size_t idx = ((size_t)b * CCH + mrow) * NPIX + n;
                outf[idx] = xres[idx] + acc[j][r] + bv[r];
            }
        }
    }
}

// ---------- flash attention: 16 queries/wave, key tiles of 32, online softmax ----
// S^T = K(A: M=key,K=c) x q^T(B: K=c,N=query)  -> row stats are in-lane + 1 shuffle
// O^T += V^T(A: M=c,K=key) x P^T(B: K=key,N=query)
// Softmax in exp2 domain (log2e folded into q); rescale skipped via wave-uniform
// ballot when no query's running max changed (alpha == 1 for every lane).
__global__ void attn_kernel(const _Float16* __restrict__ q, const _Float16* __restrict__ k,
                            const _Float16* __restrict__ vT, _Float16* __restrict__ o) {
    int lane = threadIdx.x & 31, wave = threadIdx.x >> 5;
    int b = blockIdx.y;
    int qbase = blockIdx.x * 128 + wave * 16;
    const _Float16* qp = q + ((size_t)b * NPIX + qbase) * CCH;
    const _Float16* kp = k + (size_t)b * NPIX * CCH;
    const _Float16* vp = vT + (size_t)b * CCH * NPIX;
    int hi = lane >> 4;

    v8f zero = {0.f, 0.f, 0.f, 0.f, 0.f, 0.f, 0.f, 0.f};
    v8f oacc[16];
#pragma unroll
    for (int i = 0; i < 16; ++i) oacc[i] = zero;
    float mrun = -1e30f, lrun = 0.f;

    for (int jt = 0; jt < 128; ++jt) {
        int j0 = jt * 32;
        // prefetch next K tile (16KB contiguous) into the cache hierarchy
        if (jt + 1 < 128)
            __builtin_prefetch(kp + (size_t)(j0 + 32) * CCH + (size_t)lane * 256, 0, 3);

        v8f s0 = zero, s1 = zero;
#pragma unroll
        for (int ks = 0; ks < 8; ++ks) {
            int k0 = ks * 32;
            v16h bq = load_b_frag(qp, CCH, k0, lane);
            v16h a0 = load_a_frag(kp + (size_t)j0 * CCH, CCH, k0, lane);
            v16h a1 = load_a_frag(kp + (size_t)(j0 + 16) * CCH, CCH, k0, lane);
            s0 = wmma_f16(a0, bq, s0);
            s1 = wmma_f16(a1, bq, s1);
        }
        // per-query tile max (8 keys in s0 + 8 in s1 per lane; partner has other 16)
        float tmax = -1e30f;
#pragma unroll
        for (int r = 0; r < 8; ++r) tmax = fmaxf(tmax, fmaxf(s0[r], s1[r]));
        tmax = fmaxf(tmax, __shfl_xor(tmax, 16, 32));

        float mold = mrun;
        float mn = fmaxf(mold, tmax);
        mrun = mn;

        // rescale only when some query's max actually moved (uniform branch)
        if (__ballot(tmax > mold) != 0ull) {
            float alpha = exp2f(mold - mn);
            lrun *= alpha;
#pragma unroll
            for (int ct = 0; ct < 16; ++ct)
#pragma unroll
                for (int r = 0; r < 8; ++r) oacc[ct][r] *= alpha;
        }

        float e0[8], e1[8], tsum = 0.f;
#pragma unroll
        for (int r = 0; r < 8; ++r) {
            e0[r] = exp2f(s0[r] - mn);
            e1[r] = exp2f(s1[r] - mn);
            tsum += e0[r] + e1[r];
        }
        tsum += __shfl_xor(tsum, 16, 32);
        lrun += tsum;

        // Build P^T B-fragment: lane<16 needs keys j0..j0+15 (own D0 + partner D0);
        // lane>=16 needs keys j0+16..j0+31 (partner D1 + own D1).
        v16h pb;
#pragma unroll
        for (int t = 0; t < 8; ++t) {
            float p1 = __shfl_xor(e1[t], 16, 32);
            float p0 = __shfl_xor(e0[t], 16, 32);
            pb[t]     = (_Float16)(hi ? p1 : e0[t]);
            pb[t + 8] = (_Float16)(hi ? e1[t] : p0);
        }

#pragma unroll
        for (int ct = 0; ct < 16; ++ct) {
            v16h av = load_a_frag(vp + (size_t)(ct * 16) * NPIX, NPIX, j0, lane);
            oacc[ct] = wmma_f16(av, pb, oacc[ct]);
        }
    }

    float rl = 1.0f / lrun;
    _Float16* op = o + ((size_t)b * NPIX + qbase + (lane & 15)) * CCH + hi * 8;
#pragma unroll
    for (int ct = 0; ct < 16; ++ct) {
        v8h pk;
#pragma unroll
        for (int r = 0; r < 8; ++r) pk[r] = (_Float16)(oacc[ct][r] * rl);
        *(v8h*)(op + ct * 16) = pk;
    }
}

extern "C" void kernel_launch(void* const* d_in, const int* in_sizes, int n_in,
                              void* d_out, int out_size, void* d_ws, size_t ws_size,
                              hipStream_t stream) {
    (void)in_sizes; (void)n_in; (void)out_size; (void)ws_size;
    const float* x  = (const float*)d_in[0];
    const float* gg = (const float*)d_in[1];
    const float* gb = (const float*)d_in[2];
    const float* wq = (const float*)d_in[3];
    const float* bq = (const float*)d_in[4];
    const float* wk = (const float*)d_in[5];
    const float* bk = (const float*)d_in[6];
    const float* wv = (const float*)d_in[7];
    const float* bv = (const float*)d_in[8];
    const float* wp = (const float*)d_in[9];
    const float* bp = (const float*)d_in[10];
    float* out = (float*)d_out;

    const size_t M = (size_t)2 * NPIX * CCH;   // elements per activation buffer
    _Float16* h16  = (_Float16*)d_ws;
    _Float16* q16  = h16 + M;
    _Float16* k16  = q16 + M;
    _Float16* vT16 = k16 + M;
    _Float16* o16  = vT16 + M;
    _Float16* w16  = o16 + M;                  // 4 * 256*256 f16 weights

    wconv_kernel<<<dim3(1024), dim3(256), 0, stream>>>(wq, wk, wv, wp, w16);
    gn_kernel<<<dim3(32, 2), dim3(256), 0, stream>>>(x, gg, gb, h16);
    gemm_kernel<<<dim3(16, 8, 2), dim3(256), 0, stream>>>(w16,           bq, h16, q16, nullptr, nullptr, 0);
    gemm_kernel<<<dim3(16, 8, 2), dim3(256), 0, stream>>>(w16 + 65536,   bk, h16, k16, nullptr, nullptr, 0);
    gemm_kernel<<<dim3(16, 8, 2), dim3(256), 0, stream>>>(w16 + 131072,  bv, h16, vT16, nullptr, nullptr, 1);
    attn_kernel<<<dim3(32, 2), dim3(256), 0, stream>>>(q16, k16, vT16, o16);
    gemm_kernel<<<dim3(16, 8, 2), dim3(256), 0, stream>>>(w16 + 196608,  bp, o16, nullptr, out, x, 2);
}
